// CrossAttention_2834678415812
// MI455X (gfx1250) — matrix operand order repaired and checked by reading, not compiled, not run on hardware
//
#include <hip/hip_runtime.h>

typedef __attribute__((ext_vector_type(16))) _Float16 v16h;
typedef __attribute__((ext_vector_type(8)))  float    v8f;
typedef __attribute__((ext_vector_type(4)))  float    v4f;

#define BB 2
#define LL 256
#define HH 768
#define NHEAD 12
#define DH 64

// ---------------------------------------------------------------------------
// Generic WMMA GEMM:  C = act( A(MxK) @ opB(KxN) + bias[n] + bias2[n] ) [+ resid]
//   opB[k][n] = BTRANS ? W[n*ldw + k] : W[k*ldw + n]
// BTRANS is a template parameter -> branch-free inner loop per variant.
// Batched via blockIdx.y / blockIdx.z with independent strides per operand.
// M,N multiples of 16; K multiple of 32; lda/ldw multiples of 4.
// 256 threads = 8 waves, one 16x16 tile per wave.
// ---------------------------------------------------------------------------
template <int BTRANS>
__global__ void __launch_bounds__(256)
gemm_wmma_kernel(const float* __restrict__ A, int lda, long long AsY, long long AsZ,
                 const float* __restrict__ W, int ldw, long long WsY, long long WsZ,
                 const float* __restrict__ bias, const float* __restrict__ bias2,
                 const float* __restrict__ resid,
                 float* __restrict__ C, int ldc, long long CsY, long long CsZ,
                 int M, int N, int K, int leaky)
{
    const int lane = threadIdx.x & 31;
    const int wid  = threadIdx.x >> 5;
    const int tilesN = N >> 4;
    const int tilesM = M >> 4;
    int tile = blockIdx.x * 8 + wid;
    if (tile >= tilesM * tilesN) return;            // wave-uniform exit
    const int tm = tile / tilesN;
    const int tn = tile - tm * tilesN;

    const float* Ab = A + (long long)blockIdx.y * AsY + (long long)blockIdx.z * AsZ;
    const float* Wb = W + (long long)blockIdx.y * WsY + (long long)blockIdx.z * WsZ;
    float*       Cb = C + (long long)blockIdx.y * CsY + (long long)blockIdx.z * CsZ;

    const int m0 = tm << 4, n0 = tn << 4;
    // A fragment: lane holds row m0+(lane&15); 16 consecutive k starting at (lane>>4)*16
    const int arow = m0 + (lane & 15);
    const int ahs  = (lane >> 4) << 4;              // 0 or 16
    // B fragment: lane holds col n0+(lane&15); 16 consecutive k starting at (lane>>4)*16
    const int bn   = n0 + (lane & 15);
    const int bks  = (lane >> 4) << 4;              // 0 or 16

    v8f acc = {};
    for (int k0 = 0; k0 < K; k0 += 32) {
        union { v16h v; _Float16 h[16]; } af, bf;
        // ---- A fragment: four 16B vector loads ----
        {
            const v4f* ap = (const v4f*)(Ab + (long long)arow * lda + (k0 + ahs));
            v4f a0 = ap[0], a1 = ap[1], a2 = ap[2], a3 = ap[3];   // k..k+15
#pragma unroll
            for (int q = 0; q < 4; ++q) {
                af.h[q]      = (_Float16)a0[q];
                af.h[4 + q]  = (_Float16)a1[q];
                af.h[8 + q]  = (_Float16)a2[q];
                af.h[12 + q] = (_Float16)a3[q];
            }
        }
        // ---- B fragment ----
        if (BTRANS) {                                // contiguous 64B per lane
            const v4f* wp = (const v4f*)(Wb + (long long)bn * ldw + (k0 + bks));
            v4f w0 = wp[0], w1 = wp[1], w2 = wp[2], w3 = wp[3];
#pragma unroll
            for (int q = 0; q < 4; ++q) {
                bf.h[q]      = (_Float16)w0[q];
                bf.h[4 + q]  = (_Float16)w1[q];
                bf.h[8 + q]  = (_Float16)w2[q];
                bf.h[12 + q] = (_Float16)w3[q];
            }
        } else {                                     // strided column gather
            const float* wp = Wb + (long long)(k0 + bks) * ldw + bn;
#pragma unroll
            for (int q = 0; q < 16; ++q) bf.h[q] = (_Float16)wp[(long long)q * ldw];
        }
        acc = __builtin_amdgcn_wmma_f32_16x16x32_f16(false, af.v, false, bf.v,
                                                     (short)0, acc, false, false);
    }
    // C layout: lane holds col n0+(lane&15); rows m0 + (lane>>4)*8 + r
    const int col = n0 + (lane & 15);
    const int rb  = m0 + ((lane >> 4) << 3);
    float bsum = 0.f;
    if (bias)  bsum += bias[col];
    if (bias2) bsum += bias2[col];
#pragma unroll
    for (int r = 0; r < 8; ++r) {
        const int row = rb + r;
        float v = acc[r] + bsum;
        if (leaky) v = v > 0.f ? v : 0.01f * v;
        if (resid) v += resid[(long long)row * ldc + col];
        Cb[(long long)row * ldc + col] = v;
    }
}

// ---------------------------------------------------------------------------
// c[b,h,i] = qv[b,i,h*D:]·bkr[h*D:]   (constant-over-j part of positional term)
// ---------------------------------------------------------------------------
__global__ void __launch_bounds__(256)
cbias_kernel(const float* __restrict__ qv, const float* __restrict__ bkr,
             float* __restrict__ cbuf)
{
    int idx = blockIdx.x * 256 + threadIdx.x;       // (b*NH + h)*L + i
    if (idx >= BB * NHEAD * LL) return;
    int i = idx % LL;
    int h = (idx / LL) % NHEAD;
    int b = idx / (LL * NHEAD);
    const float* qr = qv + ((long long)(b * LL + i)) * HH + h * DH;
    const float* br = bkr + h * DH;
    float s = 0.f;
#pragma unroll 8
    for (int d = 0; d < DH; ++d) s += qr[d] * br[d];
    cbuf[idx] = s;
}

// ---------------------------------------------------------------------------
// Fused positional-logit WMMA + softmax. One block per (b,i); 16 waves.
//   logits[h,j] = pos[b,i,j,:]·T[b,h,i,:]  (WMMA, f32 acc)  + attnS[b,h,i,j] + c[b,h,i]
//   then masked softmax over j; probabilities written back into attnS.
// pos_emb (402 MB) is streamed exactly once with non-temporal b128 loads and
// global_prefetch_b8 lookahead; f32->f16 conversion happens in registers.
// ---------------------------------------------------------------------------
__global__ void __launch_bounds__(512)
attn_softmax_kernel(const float* __restrict__ T,      // (B,NH,L,H)
                    const float* __restrict__ pos,    // (B,L,L,H)
                    const float* __restrict__ cbuf,   // (B,NH,L)
                    const unsigned char* __restrict__ mask, // (B,L)
                    float* __restrict__ attnS)        // (B,NH,L,L) in: content logits, out: probs
{
    __shared__ _Float16 sT[16 * HH];                  // 24 KB : T rows (h padded to 16)
    __shared__ float    sp[16 * LL];                  // 16 KB : logits / probs

    const int tid  = threadIdx.x;
    const int lane = tid & 31;
    const int wid  = tid >> 5;                        // 0..15
    const int bi   = blockIdx.x;
    const int b    = bi / LL;
    const int i    = bi - b * LL;

    // stage T[b,:,i,:] -> LDS (f16), zero-pad heads 12..15
    const float* Tbase = T + ((long long)b * NHEAD * LL + i) * HH;
    for (int idx = tid; idx < 16 * HH; idx += 512) {
        int h = idx / HH, e = idx - h * HH;
        sT[idx] = (h < NHEAD) ? (_Float16)Tbase[(long long)h * LL * HH + e]
                              : (_Float16)0.f;
    }
    __syncthreads();

    // ---- phase 1: WMMA logits, wave w owns j-tile [w*16, w*16+16) ----
    const int j0  = wid << 4;
    const int am  = lane & 15;
    const int ahs = (lane >> 4) << 4;
    const int bj  = j0 + (lane & 15);
    const int bks = (lane >> 4) << 4;
    const float* prow = pos + (((long long)b * LL + i) * LL + bj) * HH;

    v8f acc = {};
    for (int e0 = 0; e0 < HH; e0 += 32) {
        // prefetch one chunk (512 floats) ahead in the stream; speculative at tail
        __builtin_prefetch(prow + e0 + 512, 0, 1);
        union { v16h v; _Float16 h[16]; } af, bf;
        const _Float16* tp = sT + am * HH + e0 + ahs;
#pragma unroll
        for (int q = 0; q < 8; ++q) af.h[q]     = tp[q];
#pragma unroll
        for (int q = 0; q < 8; ++q) af.h[8 + q] = tp[8 + q];
        // non-temporal b128 loads of the single-use pos stream (64B aligned)
        const v4f* pp = (const v4f*)(prow + e0 + bks);
        v4f pv[4];
#pragma unroll
        for (int t = 0; t < 4; ++t) pv[t] = __builtin_nontemporal_load(pp + t);
#pragma unroll
        for (int q = 0; q < 16; ++q) bf.h[q] = (_Float16)pv[q >> 2][q & 3];
        acc = __builtin_amdgcn_wmma_f32_16x16x32_f16(false, af.v, false, bf.v,
                                                     (short)0, acc, false, false);
    }
    const int col = j0 + (lane & 15);
    const int rb  = (lane >> 4) << 3;
    const bool mok = mask[b * LL + col] != 0;
#pragma unroll
    for (int r = 0; r < 8; ++r) {
        const int h = rb + r;
        float v = 0.f;
        if (h < NHEAD) {
            v = acc[r]
              + attnS[(((long long)b * NHEAD + h) * LL + i) * LL + col]
              + cbuf[((long long)b * NHEAD + h) * LL + i];
            if (!mok) v = -1000000.f;
        }
        sp[h * LL + col] = v;
    }
    __syncthreads();

    // ---- phase 2: softmax, one wave per head row; write probs back to attnS ----
    if (wid < NHEAD) {
        const int h = wid;
        float vals[8];
        float mx = -3.4e38f;
#pragma unroll
        for (int q = 0; q < 8; ++q) {
            vals[q] = sp[h * LL + lane + q * 32];
            mx = fmaxf(mx, vals[q]);
        }
#pragma unroll
        for (int o = 16; o > 0; o >>= 1) mx = fmaxf(mx, __shfl_xor(mx, o, 32));
        float sum = 0.f;
#pragma unroll
        for (int q = 0; q < 8; ++q) { vals[q] = __expf(vals[q] - mx); sum += vals[q]; }
#pragma unroll
        for (int o = 16; o > 0; o >>= 1) sum += __shfl_xor(sum, o, 32);
        const float inv = 1.f / sum;
        float* prow_out = attnS + (((long long)b * NHEAD + h) * LL + i) * LL;
#pragma unroll
        for (int q = 0; q < 8; ++q) prow_out[lane + q * 32] = vals[q] * inv;
    }
}

// ---------------------------------------------------------------------------
// Row LayerNorm over H=768. One block (256 thr) per row; 3 elements/thread.
// ---------------------------------------------------------------------------
__global__ void __launch_bounds__(256)
layernorm_kernel(const float* __restrict__ in, const float* __restrict__ g,
                 const float* __restrict__ be, float* __restrict__ out)
{
    __shared__ float red[8];
    __shared__ float stat;
    const int row = blockIdx.x, tid = threadIdx.x;
    const int lane = tid & 31, wid = tid >> 5;
    const float* p = in + (long long)row * HH;
    float x0 = p[tid], x1 = p[tid + 256], x2 = p[tid + 512];

    float s = x0 + x1 + x2;
#pragma unroll
    for (int o = 16; o > 0; o >>= 1) s += __shfl_xor(s, o, 32);
    if (lane == 0) red[wid] = s;
    __syncthreads();
    if (tid == 0) {
        float t = 0.f;
        for (int q = 0; q < 8; ++q) t += red[q];
        stat = t * (1.f / (float)HH);
    }
    __syncthreads();
    const float mean = stat;
    const float d0 = x0 - mean, d1 = x1 - mean, d2 = x2 - mean;
    s = d0 * d0 + d1 * d1 + d2 * d2;
#pragma unroll
    for (int o = 16; o > 0; o >>= 1) s += __shfl_xor(s, o, 32);
    if (lane == 0) red[wid] = s;
    __syncthreads();
    if (tid == 0) {
        float t = 0.f;
        for (int q = 0; q < 8; ++q) t += red[q];
        stat = rsqrtf(t * (1.f / (float)HH) + 1e-5f);
    }
    __syncthreads();
    const float rs = stat;
    float* o = out + (long long)row * HH;
    o[tid]       = d0 * rs * g[tid]       + be[tid];
    o[tid + 256] = d1 * rs * g[tid + 256] + be[tid + 256];
    o[tid + 512] = d2 * rs * g[tid + 512] + be[tid + 512];
}

// ---------------------------------------------------------------------------
extern "C" void kernel_launch(void* const* d_in, const int* in_sizes, int n_in,
                              void* d_out, int out_size, void* d_ws, size_t ws_size,
                              hipStream_t stream)
{
    (void)in_sizes; (void)n_in; (void)out_size; (void)ws_size;
    const float* x    = (const float*)d_in[0];
    const float* y    = (const float*)d_in[1];
    const unsigned char* mask = (const unsigned char*)d_in[2];
    const float* pos  = (const float*)d_in[3];
    const float* Wq   = (const float*)d_in[4];
    const float* bq   = (const float*)d_in[5];
    const float* Wv   = (const float*)d_in[6];
    const float* bv   = (const float*)d_in[7];
    const float* Wkr  = (const float*)d_in[8];
    const float* bkr  = (const float*)d_in[9];
    const float* u    = (const float*)d_in[10];
    const float* vprm = (const float*)d_in[11];
    const float* Wff  = (const float*)d_in[12];
    const float* bff  = (const float*)d_in[13];
    const float* g1   = (const float*)d_in[14];
    const float* be1  = (const float*)d_in[15];
    const float* W1   = (const float*)d_in[16];
    const float* b1   = (const float*)d_in[17];
    const float* W2   = (const float*)d_in[18];
    const float* b2   = (const float*)d_in[19];
    const float* g2   = (const float*)d_in[20];
    const float* be2  = (const float*)d_in[21];
    float* out = (float*)d_out;

    const long long NT = (long long)BB * LL;            // 512 tokens
    float* ws   = (float*)d_ws;
    float* qu   = ws;  ws += NT * HH;
    float* qv   = ws;  ws += NT * HH;
    float* vmat = ws;  ws += NT * HH;
    float* Tbuf = ws;  ws += (long long)BB * NHEAD * LL * HH;
    float* cbuf = ws;  ws += (long long)BB * NHEAD * LL;
    float* attnS= ws;  ws += (long long)BB * NHEAD * LL * LL;
    float* aout = ws;  ws += NT * HH;
    float* ffb  = ws;  ws += NT * HH;
    float* ln1b = ws;  ws += NT * HH;
    float* h1   = ws;  ws += NT * 3 * HH;
    float* o3   = ws;  ws += NT * HH;

    dim3 blk(256);
    auto nblk = [](int M, int N) { return ((M / 16) * (N / 16) + 7) / 8; };
    const long long LH  = (long long)LL * HH;
    const long long LL2 = (long long)LL * LL;

    // 1) qu = x@Wq^T + bq + u_flat     2) qv = x@Wq^T + bq + vparam_flat
    gemm_wmma_kernel<1><<<dim3(nblk(512, HH)), blk, 0, stream>>>(
        x, HH, 0, 0, Wq, HH, 0, 0, bq, u, nullptr, qu, HH, 0, 0, 512, HH, HH, 0);
    gemm_wmma_kernel<1><<<dim3(nblk(512, HH)), blk, 0, stream>>>(
        x, HH, 0, 0, Wq, HH, 0, 0, bq, vprm, nullptr, qv, HH, 0, 0, 512, HH, HH, 0);
    // 3) v = y@Wv^T + bv
    gemm_wmma_kernel<1><<<dim3(nblk(512, HH)), blk, 0, stream>>>(
        y, HH, 0, 0, Wv, HH, 0, 0, bv, nullptr, nullptr, vmat, HH, 0, 0, 512, HH, HH, 0);
    // 4) T[b,h] = qv_head(LxD) @ Wkr[hD:hD+D, :]  (batched over h=Y, b=Z; K=64)
    gemm_wmma_kernel<0><<<dim3(nblk(LL, HH), NHEAD, BB), blk, 0, stream>>>(
        qv, HH, DH, LH, Wkr, HH, (long long)DH * HH, 0,
        nullptr, nullptr, nullptr, Tbuf, HH, LH, (long long)NHEAD * LH, LL, HH, DH, 0);
    // 5) c[b,h,i]
    cbias_kernel<<<dim3((BB * NHEAD * LL + 255) / 256), blk, 0, stream>>>(qv, bkr, cbuf);
    // 6) content logits: attnS[b,h] = qu_head(LxD) @ k_head(DxL)  (k = heads(y))
    gemm_wmma_kernel<1><<<dim3(nblk(LL, LL), NHEAD, BB), blk, 0, stream>>>(
        qu, HH, DH, LH, y, HH, DH, LH,
        nullptr, nullptr, nullptr, attnS, LL, LL2, (long long)NHEAD * LL2, LL, LL, DH, 0);
    // 7) fused positional WMMA + mask + softmax (probs overwrite attnS)
    attn_softmax_kernel<<<dim3(BB * LL), dim3(512), 0, stream>>>(Tbuf, pos, cbuf, mask, attnS);
    // 8) aout[b,:,hD:hD+D] = P[b,h](LxL) @ v_head(LxD)
    gemm_wmma_kernel<0><<<dim3(nblk(LL, DH), NHEAD, BB), blk, 0, stream>>>(
        attnS, LL, LL2, (long long)NHEAD * LL2, vmat, HH, DH, LH,
        nullptr, nullptr, nullptr, aout, HH, DH, LH, LL, DH, LL, 0);
    // 9) ff = leaky(aout@Wff^T + bff) + x
    gemm_wmma_kernel<1><<<dim3(nblk(512, HH)), blk, 0, stream>>>(
        aout, HH, 0, 0, Wff, HH, 0, 0, bff, nullptr, x, ffb, HH, 0, 0, 512, HH, HH, 1);
    // 10) ln1
    layernorm_kernel<<<dim3(BB * LL), blk, 0, stream>>>(ffb, g1, be1, ln1b);
    // 11) h1 = leaky(ln1@W1^T + b1)   (N = 2304)
    gemm_wmma_kernel<1><<<dim3(nblk(512, 3 * HH)), blk, 0, stream>>>(
        ln1b, HH, 0, 0, W1, HH, 0, 0, b1, nullptr, nullptr, h1, 3 * HH, 0, 0, 512, 3 * HH, HH, 1);
    // 12) o3 = h1@W2^T + b2 + ln1    (K = 2304)
    gemm_wmma_kernel<1><<<dim3(nblk(512, HH)), blk, 0, stream>>>(
        h1, 3 * HH, 0, 0, W2, 3 * HH, 0, 0, b2, nullptr, ln1b, o3, HH, 0, 0, 512, HH, 3 * HH, 0);
    // 13) final LN -> d_out
    layernorm_kernel<<<dim3(BB * LL), blk, 0, stream>>>(o3, g2, be2, out);
}
